// SeqToSeq_19705309954219
// MI455X (gfx1250) — matrix operand order repaired and checked by reading, not compiled, not run on hardware
//
#include <hip/hip_runtime.h>
#include <hip/hip_bf16.h>

// ---------------------------------------------------------------------------
// SeqToSeq LSTM + MLP head for MI455X (gfx1250), bf16 WMMA w/ f32 accumulate.
//
// All matmuls lowered to v_wmma_f32_16x16x32_bf16 (16x16x32, wave32).
// Weights converted once per launch into bf16 [N][K]-transposed layout so a
// lane's B-fragment is 16 contiguous K elements (same k order as the A fragment).
// LSTM input+recurrent kernels fused: z_t = [x_t | h] @ [K;R] + b.
//
// Round-3: 32-row batch tiles (two M-subtiles share every B fragment ->
// 2x weight-stream reuse, 8 WMMA per loop iter), raw v_rcp_f32 in the gate
// epilogue instead of IEEE division chains. Loops kept rolled to preserve
// the spill-free codegen achieved in round 2.
// ---------------------------------------------------------------------------

typedef __attribute__((ext_vector_type(16))) __bf16 v16bf;
typedef __attribute__((ext_vector_type(8)))  __bf16 v8bf;
typedef __attribute__((ext_vector_type(8)))  float  v8f;

#define BATCH   32768
#define TSTEPS  7
#define UDIM    356
#define UPAD    384             // 24 tiles of 16
#define DPAD    64              // padded input width (covers 60 and 36)
#define KC      (DPAD + UPAD)   // 448 = 14 k-steps of 32
#define KITER   (KC / 32)       // 14
#define NG      (4 * UPAD)      // 1536 rows in combined weight
#define FPAD    2496            // 7*356 = 2492 padded to /32
#define BT      32              // batch rows per block (2 M-subtiles)
#define UT_PER_WAVE 3           // 24 u-tiles / 8 waves

union V16U { v16bf v; v8bf h[2]; };

__device__ __forceinline__ v8f wmma_bf16(v16bf a, v16bf b, v8f c) {
    return __builtin_amdgcn_wmma_f32_16x16x32_bf16(
        /*neg_a=*/false, a, /*neg_b=*/false, b,
        /*c_mod=*/(short)0, c, /*reuse_a=*/false, /*reuse_b=*/false);
}

// A fragment (16x32 bf16, M x K): lane L holds row L%16, 16 contiguous K
// starting at 32*kk + 16*(L>>4), where k0 = 32*kk + 8*(L>>4).
__device__ __forceinline__ v16bf ld_frag_a(const __bf16* rowp, int k0) {
    V16U u;
    const int kb = (k0 & ~31) + 2 * (k0 & 31);
    u.h[0] = *(const v8bf*)(rowp + kb);
    u.h[1] = *(const v8bf*)(rowp + kb + 8);
    return u.v;
}

// B fragment (32x16 bf16, K x N): lane L holds col L%16, 16 contiguous K
// starting at 32*kk + 16*(L>>4), from the [N][K]-transposed weight.
__device__ __forceinline__ v16bf ld_frag_b(const __bf16* p) {
    V16U u;
    u.h[0] = *(const v8bf*)(p);
    u.h[1] = *(const v8bf*)(p + 8);
    return u.v;
}

__device__ __forceinline__ float fast_rcp(float x) {
    return __builtin_amdgcn_rcpf(x);       // raw v_rcp_f32, no div chain
}
__device__ __forceinline__ float sigf(float x) {
    float e = __expf(-x);
    return fast_rcp(1.f + e);
}
__device__ __forceinline__ float tanhf_fast(float x) {
    x = fminf(fmaxf(x, -15.f), 15.f);
    float e = __expf(2.f * x);
    return (e - 1.f) * fast_rcp(e + 1.f);
}

// ---------------------------------------------------------------------------
// Weight prep: combined LSTM weight, transposed bf16:
// dst[(q*UPAD+u)*KC + k]:  k<DPAD -> input kernel K[k][q*356+u]
//                          k>=DPAD -> recurrent R[k-DPAD][q*356+u]
// ---------------------------------------------------------------------------
__global__ void build_lstm_wt(const float* __restrict__ Kin,   // [D][1424]
                              const float* __restrict__ Rin,   // [356][1424]
                              int D, __bf16* __restrict__ dst) {
    int idx = blockIdx.x * blockDim.x + threadIdx.x;
    if (idx >= NG * KC) return;
    int row = idx / KC, k = idx % KC;
    int q = row / UPAD, u = row % UPAD;
    float v = 0.f;
    if (u < UDIM) {
        int n = q * UDIM + u;
        if (k < DPAD) {
            if (k < D) v = Kin[k * (4 * UDIM) + n];
        } else {
            int ku = k - DPAD;
            if (ku < UDIM) v = Rin[ku * (4 * UDIM) + n];
        }
    }
    dst[idx] = (__bf16)v;
}

// src [K][N] fp32 row-major  ->  dst [Np][Kp] bf16 (transposed, zero-padded)
__global__ void transpose_to_bf16(const float* __restrict__ src,
                                  __bf16* __restrict__ dst,
                                  int K, int N, int Kp, int Np) {
    int idx = blockIdx.x * blockDim.x + threadIdx.x;
    if (idx >= Np * Kp) return;
    int n = idx / Kp, k = idx % Kp;
    float v = (n < N && k < K) ? src[(long)k * N + n] : 0.f;
    dst[idx] = (__bf16)v;
}

// ---------------------------------------------------------------------------
// Fused LSTM: block = 32 batch rows, 256 threads = 8 waves. [x_t|h] A-tile and
// cell state c persist in LDS across all T steps. Each wave owns 3 u-tiles and
// computes all 4 gate tiles for both 16-row M-subtiles (B fragments reused 2x).
// Phase 1 (per step): accumulate gates per u-tile (8 accs live), gate math,
// stash h in regs. Barrier. Phase 2: commit h to A-tile + stage x_{t+1}.
// mode 0 = encoder (emit final h,c), mode 1 = decoder (emit h each step to F).
// ---------------------------------------------------------------------------
__global__ void __launch_bounds__(256)
lstm_fused(const float* __restrict__ x,      // [B][T][D]
           int D,
           const __bf16* __restrict__ Wt,    // [NG][KC]
           const float*  __restrict__ bias,  // [4*U]
           const __bf16* __restrict__ h0,    // [B][UPAD] (dec) or null
           const float*  __restrict__ c0,    // [B][UPAD] (dec) or null
           __bf16* __restrict__ hT,          // [B][UPAD] (enc)
           float*  __restrict__ cT,          // [B][UPAD] (enc)
           __bf16* __restrict__ F,           // [B][FPAD] (dec)
           int mode) {
    __shared__ __bf16 As[BT][KC];     // [x_t | h], bf16          (28 KB)
    __shared__ float  Cs[BT][UPAD];   // cell state, f32          (48 KB)

    const int tid  = threadIdx.x;
    const int lane = tid & 31;
    const int wave = tid >> 5;
    const int rsel = lane >> 4;       // 0/1: lane half selects row/K sub-chunk
    const int cn   = lane & 15;
    const long b0  = (long)blockIdx.x * BT;
    const long GQ  = (long)UPAD * KC; // weight stride between gate blocks

    // ---- init persistent LDS state + stage x_0 --------------------------
    for (int i = tid; i < BT * KC; i += 256) {
        int r = i / KC, k = i % KC;
        __bf16 v = (__bf16)0.f;
        if (mode == 1 && k >= DPAD) v = h0[(b0 + r) * UPAD + (k - DPAD)];
        As[r][k] = v;
    }
    for (int i = tid; i < BT * UPAD; i += 256) {
        int r = i / UPAD, u = i % UPAD;
        Cs[r][u] = (mode == 1) ? c0[(b0 + r) * UPAD + u] : 0.f;
    }
    __syncthreads();
    for (int i = tid; i < BT * D; i += 256) {
        int r = i / D, d = i % D;
        As[r][d] = (__bf16)x[(b0 + r) * (long)(TSTEPS * D) + d];
    }
    __syncthreads();

    #pragma unroll 1
    for (int t = 0; t < TSTEPS; ++t) {
        __bf16 hsave[UT_PER_WAVE][2][8];   // per-lane h results, in regs

        #pragma unroll 1
        for (int j = 0; j < UT_PER_WAVE; ++j) {
            const int u = (wave * UT_PER_WAVE + j) * 16 + cn;

            v8f acc[2][4];
            #pragma unroll
            for (int q = 0; q < 4; ++q) {
                float bv = (u < UDIM) ? bias[q * UDIM + u] : 0.f;
                acc[0][q] = (v8f){bv, bv, bv, bv, bv, bv, bv, bv};
                acc[1][q] = acc[0][q];
            }

            const __bf16* bbase = Wt + (long)u * KC + rsel * 16;
            const __bf16* arow0 = &As[cn][0];
            const __bf16* arow1 = &As[16 + cn][0];
            #pragma unroll 1
            for (int kk = 0; kk < KITER; ++kk) {
                v16bf a0 = ld_frag_a(arow0, kk * 32 + rsel * 8);
                v16bf a1 = ld_frag_a(arow1, kk * 32 + rsel * 8);
                v16bf w0 = ld_frag_b(bbase + 0 * GQ + kk * 32);
                v16bf w1 = ld_frag_b(bbase + 1 * GQ + kk * 32);
                v16bf w2 = ld_frag_b(bbase + 2 * GQ + kk * 32);
                v16bf w3 = ld_frag_b(bbase + 3 * GQ + kk * 32);
                acc[0][0] = wmma_bf16(a0, w0, acc[0][0]);
                acc[1][0] = wmma_bf16(a1, w0, acc[1][0]);
                acc[0][1] = wmma_bf16(a0, w1, acc[0][1]);
                acc[1][1] = wmma_bf16(a1, w1, acc[1][1]);
                acc[0][2] = wmma_bf16(a0, w2, acc[0][2]);
                acc[1][2] = wmma_bf16(a1, w2, acc[1][2]);
                acc[0][3] = wmma_bf16(a0, w3, acc[0][3]);
                acc[1][3] = wmma_bf16(a1, w3, acc[1][3]);
            }

            // gate math + cell update (Cs columns are wave-exclusive)
            #pragma unroll
            for (int m = 0; m < 2; ++m) {
                #pragma unroll
                for (int r = 0; r < 8; ++r) {
                    int row = m * 16 + r + rsel * 8;
                    float iv = acc[m][0][r];
                    float fv = acc[m][1][r];
                    float gv = acc[m][2][r];
                    float ov = acc[m][3][r];
                    float cold = Cs[row][u];
                    float cnew = sigf(fv) * cold + sigf(iv) * tanhf_fast(gv);
                    float hv   = sigf(ov) * tanhf_fast(cnew);
                    Cs[row][u] = cnew;
                    hsave[j][m][r] = (__bf16)hv;
                    if (mode == 1 && u < UDIM)
                        F[(b0 + row) * (long)FPAD + t * UDIM + u] = (__bf16)hv;
                }
            }
        }
        __syncthreads();   // every wave finished reading As for this step

        // commit h into the A-tile; stage x_{t+1}
        #pragma unroll
        for (int j = 0; j < UT_PER_WAVE; ++j) {
            const int u = (wave * UT_PER_WAVE + j) * 16 + cn;
            #pragma unroll
            for (int m = 0; m < 2; ++m)
                #pragma unroll
                for (int r = 0; r < 8; ++r)
                    As[m * 16 + r + rsel * 8][DPAD + u] = hsave[j][m][r];
        }
        if (t + 1 < TSTEPS) {
            for (int i = tid; i < BT * D; i += 256) {
                int r = i / D, d = i % D;
                As[r][d] = (__bf16)x[(b0 + r) * (long)(TSTEPS * D)
                                     + (long)(t + 1) * D + d];
            }
        }
        __syncthreads();
    }

    if (mode == 0) {
        for (int i = tid; i < BT * UPAD; i += 256) {
            int r = i / UPAD, u = i % UPAD;
            hT[(b0 + r) * UPAD + u] = As[r][DPAD + u];
            cT[(b0 + r) * UPAD + u] = Cs[r][u];
        }
    }
}

// ---------------------------------------------------------------------------
// Generic bf16 WMMA GEMM: C[M,N] = act(A[M,Kp] @ Bt[Np,Kp]^T + bias).
// Block = 32 rows x (8 waves * 16 cols); each wave does two 16-row M-tiles
// sharing every B fragment. act: 0=none(f32 out), 1=relu(bf16), 2=tanh(bf16).
// ---------------------------------------------------------------------------
__global__ void __launch_bounds__(256)
gemm_bf16(const __bf16* __restrict__ A,
          const __bf16* __restrict__ Bt,
          const float*  __restrict__ bias,
          void* __restrict__ Cout,
          int Kp, int Nvalid, int NtileTotal, int ldo, int act) {
    const int lane = threadIdx.x & 31;
    const int wave = threadIdx.x >> 5;
    const int nt   = blockIdx.y * 8 + wave;
    if (nt >= NtileTotal) return;    // whole-wave exit: no partial-EXEC WMMA
    const int rsel = lane >> 4;
    const int cn   = lane & 15;
    const long row0 = (long)blockIdx.x * 32;
    const int col  = nt * 16 + cn;

    float bv = (col < Nvalid) ? bias[col] : 0.f;
    v8f acc0 = (v8f){bv, bv, bv, bv, bv, bv, bv, bv};
    v8f acc1 = acc0;

    const __bf16* arow0 = A + (row0 + cn) * (long)Kp;
    const __bf16* arow1 = arow0 + 16 * (long)Kp;
    const __bf16* brow  = Bt + (long)(nt * 16 + cn) * Kp + rsel * 16;
    #pragma unroll 1
    for (int kk = 0; kk < Kp / 32; ++kk) {
        v16bf a0 = ld_frag_a(arow0, kk * 32 + rsel * 8);
        v16bf a1 = ld_frag_a(arow1, kk * 32 + rsel * 8);
        v16bf b  = ld_frag_b(brow + kk * 32);
        acc0 = wmma_bf16(a0, b, acc0);
        acc1 = wmma_bf16(a1, b, acc1);
    }

    #pragma unroll
    for (int m = 0; m < 2; ++m) {
        #pragma unroll
        for (int r = 0; r < 8; ++r) {
            long row = row0 + m * 16 + r + rsel * 8;
            if (col < Nvalid) {
                float v = (m == 0) ? acc0[r] : acc1[r];
                if (act == 1)      v = fmaxf(v, 0.f);
                else if (act == 2) v = tanhf_fast(v);
                if (act == 0) ((float*)Cout)[row * ldo + col] = v;
                else          ((__bf16*)Cout)[row * ldo + col] = (__bf16)v;
            }
        }
    }
}

// ---------------------------------------------------------------------------
// Host-side orchestration
// ---------------------------------------------------------------------------
extern "C" void kernel_launch(void* const* d_in, const int* in_sizes, int n_in,
                              void* d_out, int out_size, void* d_ws, size_t ws_size,
                              hipStream_t stream) {
    const float* x_batch = (const float*)d_in[0];
    const float* m_batch = (const float*)d_in[1];
    const float* enc_k   = (const float*)d_in[2];
    const float* enc_r   = (const float*)d_in[3];
    const float* enc_b   = (const float*)d_in[4];
    const float* dec_k   = (const float*)d_in[5];
    const float* dec_r   = (const float*)d_in[6];
    const float* dec_b   = (const float*)d_in[7];
    const float* W_map   = (const float*)d_in[8];
    const float* b_map   = (const float*)d_in[9];
    const float* W_d1    = (const float*)d_in[10];
    const float* b_d1    = (const float*)d_in[11];
    const float* W_d2    = (const float*)d_in[12];
    const float* b_d2    = (const float*)d_in[13];
    const float* W_out   = (const float*)d_in[14];
    const float* b_out   = (const float*)d_in[15];

    // ---- workspace carve-up (all chunks 256B-aligned by construction) ----
    const size_t SZ_WLSTM = (size_t)NG * KC * 2;            //  1.38 MB each
    const size_t SZ_WMAP  = (size_t)768 * FPAD * 2;         //  3.83 MB
    const size_t SZ_WD    = (size_t)768 * 768 * 2;          //  1.18 MB each
    const size_t SZ_WOUT  = (size_t)176 * 768 * 2;          //  0.27 MB
    const size_t SZ_H     = (size_t)BATCH * UPAD * 2;       // 25.2 MB
    const size_t SZ_C     = (size_t)BATCH * UPAD * 4;       // 50.3 MB
    const size_t SZ_F     = (size_t)BATCH * FPAD * 2;       // 163.6 MB
    const size_t SZ_ACT   = (size_t)BATCH * 768 * 2;        // 50.3 MB each

    char* ws = (char*)d_ws;
    size_t off = 0;
    auto carve = [&](size_t bytes) { void* p = ws + off; off += bytes; return p; };
    __bf16* WencT = (__bf16*)carve(SZ_WLSTM);
    __bf16* WdecT = (__bf16*)carve(SZ_WLSTM);
    __bf16* WmapT = (__bf16*)carve(SZ_WMAP);
    __bf16* Wd1T  = (__bf16*)carve(SZ_WD);
    __bf16* Wd2T  = (__bf16*)carve(SZ_WD);
    __bf16* WoutT = (__bf16*)carve(SZ_WOUT);
    __bf16* hEnc  = (__bf16*)carve(SZ_H);
    float*  cEnc  = (float*) carve(SZ_C);
    __bf16* Fbuf  = (__bf16*)carve(SZ_F);
    __bf16* M1    = (__bf16*)carve(SZ_ACT);
    __bf16* D1    = (__bf16*)carve(SZ_ACT);
    __bf16* D2    = (__bf16*)carve(SZ_ACT);
    (void)in_sizes; (void)n_in; (void)out_size; (void)ws_size;

    // ---- weight prep ----------------------------------------------------
    {
        int tot = NG * KC;
        build_lstm_wt<<<(tot + 255) / 256, 256, 0, stream>>>(enc_k, enc_r, 60, WencT);
        build_lstm_wt<<<(tot + 255) / 256, 256, 0, stream>>>(dec_k, dec_r, 36, WdecT);
    }
    {
        int tot = 768 * FPAD;
        transpose_to_bf16<<<(tot + 255) / 256, 256, 0, stream>>>(W_map, WmapT, 2492, 768, FPAD, 768);
    }
    {
        int tot = 768 * 768;
        transpose_to_bf16<<<(tot + 255) / 256, 256, 0, stream>>>(W_d1, Wd1T, 768, 768, 768, 768);
        transpose_to_bf16<<<(tot + 255) / 256, 256, 0, stream>>>(W_d2, Wd2T, 768, 768, 768, 768);
    }
    {
        int tot = 176 * 768;
        transpose_to_bf16<<<(tot + 255) / 256, 256, 0, stream>>>(W_out, WoutT, 768, 168, 768, 176);
    }

    // ---- encoder / decoder LSTMs ----------------------------------------
    const int nLstmBlocks = BATCH / BT;   // 1024
    lstm_fused<<<nLstmBlocks, 256, 0, stream>>>(
        x_batch, 60, WencT, enc_b, nullptr, nullptr, hEnc, cEnc, nullptr, 0);
    lstm_fused<<<nLstmBlocks, 256, 0, stream>>>(
        m_batch, 36, WdecT, dec_b, hEnc, cEnc, nullptr, nullptr, Fbuf, 1);

    // ---- MLP head -------------------------------------------------------
    const int gx = BATCH / 32;            // 1024
    gemm_bf16<<<dim3(gx, 6), 256, 0, stream>>>(Fbuf, WmapT, b_map, M1,
                                               FPAD, 768, 48, 768, /*relu*/1);
    gemm_bf16<<<dim3(gx, 6), 256, 0, stream>>>(M1, Wd1T, b_d1, D1,
                                               768, 768, 48, 768, /*tanh*/2);
    gemm_bf16<<<dim3(gx, 6), 256, 0, stream>>>(D1, Wd2T, b_d2, D2,
                                               768, 768, 48, 768, /*tanh*/2);
    gemm_bf16<<<dim3(gx, 2), 256, 0, stream>>>(D2, WoutT, b_out, d_out,
                                               768, 168, 11, 168, /*none*/0);
}
